// AttentionBlock_26388279067243
// MI455X (gfx1250) — matrix operand order repaired and checked by reading, not compiled, not run on hardware
//
#include <hip/hip_runtime.h>
#include <hip/hip_bf16.h>

// ---------------------------------------------------------------------------
// AttentionBlock for MI455X (gfx1250, wave32, WMMA f32_16x16x32_f16)
// B=8, C=512, T=1024, heads=8, ch=64, groups=32
// ---------------------------------------------------------------------------

#define B_   8
#define C_   512
#define T_   1024
#define NH_  8
#define CH_  64
#define G_   32
#define CPG_ 16            // channels per group
#define EPS_ 1e-5f
#define SCALE_ 0.59460355750136053f   // 1/sqrt(sqrt(8))

typedef __attribute__((ext_vector_type(16))) _Float16 v16h;
typedef __attribute__((ext_vector_type(8)))  _Float16 v8h;
typedef __attribute__((ext_vector_type(8)))  float    v8f;
typedef __attribute__((ext_vector_type(4)))  int      v4i;

// ---- CDNA5 async global->LDS path (ASYNCcnt), with sync fallback ----------
#if defined(__has_builtin)
#if __has_builtin(__builtin_amdgcn_global_load_async_to_lds_b128) && \
    __has_builtin(__builtin_amdgcn_s_wait_asynccnt)
#define USE_ASYNC_LDS 1
#endif
#endif

#ifdef USE_ASYNC_LDS
__device__ __forceinline__ void async_cp16(const _Float16* g, _Float16* l) {
    auto gp = (__attribute__((address_space(1))) v4i*)(g);   // global src
    auto lp = (__attribute__((address_space(3))) v4i*)(l);   // LDS dst
    __builtin_amdgcn_global_load_async_to_lds_b128(gp, lp, 0, 0);
}
#endif

// Stage a 64-row x 32-half tile (row stride ld halfs) into LDS, row-major.
// 256 threads, 16 bytes each. Caller must barrier after stage_wait().
__device__ __forceinline__ void stage_tile64x32(_Float16* dst,
                                                const _Float16* src, int ld) {
    int tid = threadIdx.x;
    int row = tid >> 2, ch = (tid & 3) * 8;
#ifdef USE_ASYNC_LDS
    async_cp16(src + (size_t)row * ld + ch, dst + tid * 8);
#else
    *(v8h*)(dst + tid * 8) = *(const v8h*)(src + (size_t)row * ld + ch);
#endif
}

__device__ __forceinline__ void stage_wait() {
#ifdef USE_ASYNC_LDS
    __builtin_amdgcn_s_wait_asynccnt(0);
#endif
}

__device__ __forceinline__ v8f wmma16x16x32(v16h a, v16h b, v8f c) {
    // D = A(16x32 f16) x B(32x16 f16) + C(16x16 f32)
    return __builtin_amdgcn_wmma_f32_16x16x32_f16(
        /*neg_a=*/false, a, /*neg_b=*/false, b,
        /*c_mod=*/(short)0, c, /*reuse_a=*/false, /*reuse_b=*/false);
}

// ---- Fragment loaders, layouts per CDNA5 ISA §7.12.2 -----------
// A 16x32 f16, storage row-major [M][K], row stride ld (halfs).
// lane<16 : M=lane,    K = {0..7, 16..23}
// lane>=16: M=lane-16, K = {8..15, 24..31}
__device__ __forceinline__ v16h ldA16(const _Float16* base, int ld) {
    int lane = threadIdx.x & 31;
    int m  = lane & 15;
    int k0 = (lane >> 4) << 3;                 // 0 or 8
    const _Float16* p = base + m * ld + k0;
    v8h lo = *(const v8h*)(p);                 // K k0..k0+7
    v8h hi = *(const v8h*)(p + 16);            // K k0+16..k0+23
    v16h r;
#pragma unroll
    for (int i = 0; i < 8; ++i) { r[i] = lo[i]; r[i + 8] = hi[i]; }
    return r;
}

// A 16x32, storage row-major f32 [M][K] (convert f32 -> f16 at load).
__device__ __forceinline__ v16h ldA32(const float* base, int ld) {
    int lane = threadIdx.x & 31;
    int m  = lane & 15;
    int k0 = (lane >> 4) << 3;
    const float* p = base + m * ld + k0;
    v16h r;
#pragma unroll
    for (int i = 0; i < 8; ++i) {
        r[i]     = (_Float16)p[i];
        r[i + 8] = (_Float16)p[16 + i];
    }
    return r;
}

// B 32x16 f16 (KxN), storage N-major [N][K], row stride ld (halfs).
// lane<16 : N=lane,    K = {0..7, 16..23}
// lane>=16: N=lane-16, K = {8..15, 24..31}
__device__ __forceinline__ v16h ldB16(const _Float16* baseT, int ld) {
    int lane = threadIdx.x & 31;
    int n  = lane & 15;
    int k0 = (lane >> 4) << 3;                 // 0 or 8
    const _Float16* p = baseT + n * ld + k0;
    v8h lo = *(const v8h*)(p);
    v8h hi = *(const v8h*)(p + 16);
    v16h r;
#pragma unroll
    for (int i = 0; i < 8; ++i) { r[i] = lo[i]; r[i + 8] = hi[i]; }
    return r;
}

// C/D 16x16 f32 element mapping: elem r -> M = r + (lane>=16 ? 8:0), N = lane&15
__device__ __forceinline__ int cfragM(int r) {
    return r + (((threadIdx.x & 31) >> 4) << 3);
}
__device__ __forceinline__ int cfragN() { return threadIdx.x & 15; }

// ---------------------------------------------------------------------------
// Kernel 1: GroupNorm -> hT[b][t][c] (f16, transposed for GEMM B-fragments)
// grid = B*G blocks, 256 threads
// ---------------------------------------------------------------------------
__global__ void gn_kernel(const float* __restrict__ x,
                          const float* __restrict__ gnw,
                          const float* __restrict__ gnb,
                          _Float16* __restrict__ hT) {
    int bg = blockIdx.x;
    int b = bg >> 5, g = bg & 31;
    const float* xp = x + ((size_t)b * C_ + g * CPG_) * T_;   // 16 x 1024 block
    const int N = CPG_ * T_;                                  // 16384

    float s = 0.f, s2 = 0.f;
    for (int i = threadIdx.x; i < N; i += 256) {
        float v = xp[i];
        s += v; s2 += v * v;
    }
    // wave32 reduce then cross-wave
    __shared__ float red[16];
#pragma unroll
    for (int off = 16; off > 0; off >>= 1) {
        s  += __shfl_xor(s, off);
        s2 += __shfl_xor(s2, off);
    }
    int wave = threadIdx.x >> 5, lane = threadIdx.x & 31;
    if (lane == 0) { red[wave] = s; red[8 + wave] = s2; }
    __syncthreads();
    float S = 0.f, S2 = 0.f;
#pragma unroll
    for (int w = 0; w < 8; ++w) { S += red[w]; S2 += red[8 + w]; }
    float mean = S * (1.0f / N);
    float var  = S2 * (1.0f / N) - mean * mean;
    float inv  = rsqrtf(var + EPS_);

    for (int i = threadIdx.x; i < N; i += 256) {
        int cl = i >> 10;          // channel within group
        int t  = i & 1023;
        int c  = g * CPG_ + cl;
        float v = (xp[i] - mean) * inv * gnw[c] + gnb[c];
        hT[((size_t)b * T_ + t) * C_ + c] = (_Float16)v;
    }
}

// ---------------------------------------------------------------------------
// Kernel 2: QKV GEMM. out[o,t] = sum_c w_qkv[o,c]*h[b,c,t] + b_qkv[o]
// Block: 256 thr (8 waves), tile 128(o) x 64(t); each wave one 16-row M strip.
// B tile staged once per K-step through LDS (async-to-LDS when available).
// Epilogue routes into per-head layouts:
//   qT[bh][t][cc] *= SCALE, kT[bh][t][cc] *= SCALE, vS[bh][cc][t]
// grid = (T/64, 3C/128, B)
// ---------------------------------------------------------------------------
__global__ void qkv_kernel(const _Float16* __restrict__ hT,
                           const float* __restrict__ wqkv,
                           const float* __restrict__ bqkv,
                           _Float16* __restrict__ qT,
                           _Float16* __restrict__ kT,
                           _Float16* __restrict__ vS) {
    __shared__ _Float16 shB[64 * 32];
    int wave = threadIdx.x >> 5;
    int b  = blockIdx.z;
    int o0 = blockIdx.y * 128 + wave * 16;
    int t0 = blockIdx.x * 64;
    const _Float16* hb = hT + (size_t)b * T_ * C_;

    v8f acc[4] = {{}, {}, {}, {}};
    for (int kk = 0; kk < C_; kk += 32) {
        stage_tile64x32(shB, hb + (size_t)t0 * C_ + kk, C_);
        __builtin_prefetch(wqkv + (size_t)o0 * C_ + kk + 32, 0, 3);
        stage_wait();
        __syncthreads();
        v16h a = ldA32(wqkv + (size_t)o0 * C_ + kk, C_);
#pragma unroll
        for (int nt = 0; nt < 4; ++nt) {
            v16h bb = ldB16(shB + nt * 16 * 32, 32);
            acc[nt] = wmma16x16x32(a, bb, acc[nt]);
        }
        __syncthreads();       // done reading tile before next overwrite
    }
    // epilogue: bias + route
#pragma unroll
    for (int nt = 0; nt < 4; ++nt) {
#pragma unroll
        for (int r = 0; r < 8; ++r) {
            int o = o0 + cfragM(r);
            int t = t0 + nt * 16 + cfragN();
            float v = acc[nt][r] + bqkv[o];
            int which = o >> 9;                  // 0:q 1:k 2:v
            int ol = o & 511;
            int head = ol >> 6, cc = ol & 63;
            size_t bh = (size_t)b * NH_ + head;
            if (which == 0)
                qT[(bh * T_ + t) * CH_ + cc] = (_Float16)(v * SCALE_);
            else if (which == 1)
                kT[(bh * T_ + t) * CH_ + cc] = (_Float16)(v * SCALE_);
            else
                vS[(bh * CH_ + cc) * T_ + t] = (_Float16)v;
        }
    }
}

// ---------------------------------------------------------------------------
// Kernel 3: attention for one (head, 16-query-row) block.
// Phase 1: S[16][1024] = qT * kT^T  (scores in 64KB dynamic LDS, f32)
// Phase 2: exact f32 softmax per row (wave32 shuffle reductions)
// Phase 3: A = softmax(S) x V, K split over the 8 waves, LDS cross-wave reduce
// Output aT[b][t][c_global] (f16), ready as B-fragments for the proj GEMM.
// grid = (T/16, B*NH), 256 threads, 65536 B dynamic LDS
// ---------------------------------------------------------------------------
__global__ void attn_kernel(const _Float16* __restrict__ qT,
                            const _Float16* __restrict__ kT,
                            const _Float16* __restrict__ vS,
                            _Float16* __restrict__ aT) {
    extern __shared__ float smem[];            // 16*1024 f32 scores
    int wave = threadIdx.x >> 5;
    int lane = threadIdx.x & 31;
    int hd = blockIdx.y;                       // b*8 + head
    int b = hd >> 3, head = hd & 7;
    int t0 = blockIdx.x * 16;

    const _Float16* qbase = qT + ((size_t)hd * T_ + t0) * CH_;
    const _Float16* kbase = kT + (size_t)hd * T_ * CH_;
    const _Float16* vbase = vS + (size_t)hd * CH_ * T_;

    // ---- Phase 1: scores ----
    v16h aq0 = ldA16(qbase + 0, CH_);
    v16h aq1 = ldA16(qbase + 32, CH_);
#pragma unroll
    for (int j = 0; j < 8; ++j) {
        int s0 = (wave * 8 + j) * 16;
        v8f c = {};
        v16h b0 = ldB16(kbase + (size_t)s0 * CH_ + 0, CH_);
        c = wmma16x16x32(aq0, b0, c);
        v16h b1 = ldB16(kbase + (size_t)s0 * CH_ + 32, CH_);
        c = wmma16x16x32(aq1, b1, c);
#pragma unroll
        for (int r = 0; r < 8; ++r)
            smem[cfragM(r) * T_ + s0 + cfragN()] = c[r];
    }
    __syncthreads();

    // ---- Phase 2: softmax (2 rows per wave) ----
#pragma unroll
    for (int rr = 0; rr < 2; ++rr) {
        float* row = smem + (wave * 2 + rr) * T_;
        float mx = -3.0e38f;
        for (int i = lane; i < T_; i += 32) mx = fmaxf(mx, row[i]);
#pragma unroll
        for (int off = 16; off > 0; off >>= 1) mx = fmaxf(mx, __shfl_xor(mx, off));
        float sum = 0.f;
        for (int i = lane; i < T_; i += 32) {
            float e = __expf(row[i] - mx);
            row[i] = e;
            sum += e;
        }
#pragma unroll
        for (int off = 16; off > 0; off >>= 1) sum += __shfl_xor(sum, off);
        float inv = 1.0f / sum;
        for (int i = lane; i < T_; i += 32) row[i] *= inv;
    }
    __syncthreads();

    // ---- Phase 3: A = P x V, K(=s) split across waves ----
    v8f acc[4] = {{}, {}, {}, {}};
    int kb = wave * 128;
#pragma unroll
    for (int ks = 0; ks < 4; ++ks) {
        int kk = kb + ks * 32;
        v16h a = ldA32(smem + kk, T_);          // probs f32 -> f16
#pragma unroll
        for (int nt = 0; nt < 4; ++nt) {
            v16h bb = ldB16(vbase + (size_t)(nt * 16) * T_ + kk, T_);
            acc[nt] = wmma16x16x32(a, bb, acc[nt]);
        }
    }
    __syncthreads();                            // done reading scores
    // write partials: P[wave][m][cc]
#pragma unroll
    for (int nt = 0; nt < 4; ++nt)
#pragma unroll
        for (int r = 0; r < 8; ++r)
            smem[wave * 1024 + cfragM(r) * 64 + nt * 16 + cfragN()] = acc[nt][r];
    __syncthreads();
    // cross-wave reduction + store
    for (int i = threadIdx.x; i < 1024; i += 256) {
        int m = i >> 6, cc = i & 63;
        float s = 0.f;
#pragma unroll
        for (int w = 0; w < 8; ++w) s += smem[w * 1024 + i];
        aT[((size_t)b * T_ + t0 + m) * C_ + head * CH_ + cc] = (_Float16)s;
    }
}

// ---------------------------------------------------------------------------
// Kernel 4: out = x + w_proj @ a + b_proj   (f32 output)
// Block: 256 thr (8 waves), tile 128(o) x 64(t); B tile staged through LDS.
// grid = (T/64, C/128, B)
// ---------------------------------------------------------------------------
__global__ void proj_kernel(const _Float16* __restrict__ aT,
                            const float* __restrict__ wproj,
                            const float* __restrict__ bproj,
                            const float* __restrict__ x,
                            float* __restrict__ out) {
    __shared__ _Float16 shB[64 * 32];
    int wave = threadIdx.x >> 5;
    int b  = blockIdx.z;
    int o0 = blockIdx.y * 128 + wave * 16;
    int t0 = blockIdx.x * 64;
    const _Float16* ab = aT + (size_t)b * T_ * C_;

    v8f acc[4] = {{}, {}, {}, {}};
    for (int kk = 0; kk < C_; kk += 32) {
        stage_tile64x32(shB, ab + (size_t)t0 * C_ + kk, C_);
        __builtin_prefetch(wproj + (size_t)o0 * C_ + kk + 32, 0, 3);
        stage_wait();
        __syncthreads();
        v16h a = ldA32(wproj + (size_t)o0 * C_ + kk, C_);
#pragma unroll
        for (int nt = 0; nt < 4; ++nt) {
            v16h bb = ldB16(shB + nt * 16 * 32, 32);
            acc[nt] = wmma16x16x32(a, bb, acc[nt]);
        }
        __syncthreads();
    }
#pragma unroll
    for (int nt = 0; nt < 4; ++nt) {
#pragma unroll
        for (int r = 0; r < 8; ++r) {
            int o = o0 + cfragM(r);
            int t = t0 + nt * 16 + cfragN();
            size_t idx = ((size_t)b * C_ + o) * T_ + t;
            out[idx] = acc[nt][r] + bproj[o] + x[idx];
        }
    }
}

// ---------------------------------------------------------------------------
// Host launcher
// inputs: x, gn_weight, gn_bias, w_qkv, b_qkv, w_proj, b_proj (all f32)
// workspace layout (bytes): hT @0 (8MB), qT @8MB, kT @16MB, vS @24MB, aT @32MB
// ---------------------------------------------------------------------------
extern "C" void kernel_launch(void* const* d_in, const int* in_sizes, int n_in,
                              void* d_out, int out_size, void* d_ws, size_t ws_size,
                              hipStream_t stream) {
    const float* x     = (const float*)d_in[0];
    const float* gnw   = (const float*)d_in[1];
    const float* gnb   = (const float*)d_in[2];
    const float* wqkv  = (const float*)d_in[3];
    const float* bqkv  = (const float*)d_in[4];
    const float* wproj = (const float*)d_in[5];
    const float* bproj = (const float*)d_in[6];
    float* out = (float*)d_out;

    char* ws = (char*)d_ws;
    _Float16* hT = (_Float16*)(ws);
    _Float16* qT = (_Float16*)(ws + (size_t)8  * 1024 * 1024);
    _Float16* kT = (_Float16*)(ws + (size_t)16 * 1024 * 1024);
    _Float16* vS = (_Float16*)(ws + (size_t)24 * 1024 * 1024);
    _Float16* aT = (_Float16*)(ws + (size_t)32 * 1024 * 1024);

    gn_kernel<<<B_ * G_, 256, 0, stream>>>(x, gnw, gnb, hT);
    qkv_kernel<<<dim3(T_ / 64, (3 * C_) / 128, B_), 256, 0, stream>>>(
        hT, wqkv, bqkv, qT, kT, vS);
    attn_kernel<<<dim3(T_ / 16, B_ * NH_), 256, 16 * T_ * sizeof(float), stream>>>(
        qT, kT, vS, aT);
    proj_kernel<<<dim3(T_ / 64, C_ / 128, B_), 256, 0, stream>>>(
        aT, wproj, bproj, x, out);
}
